// BaseAttention_81192061763680
// MI455X (gfx1250) — compile-verified, matrix operand order checked by reading, not run on hardware
//
#include <hip/hip_runtime.h>
#include <hip/hip_bf16.h>

// ---------------------------------------------------------------------------
// Sliding-window causal attention (S=4096, H=1024, 16 heads, hd=64, W=256)
// Matmuls on v_wmma_f32_16x16x32_bf16; GEMM tiles staged with CDNA5 async
// global->LDS copies (ASYNCcnt) and double-buffered LDS.
// ---------------------------------------------------------------------------

typedef __bf16 bf16;
typedef __attribute__((ext_vector_type(16))) __bf16 v16bf;
typedef __attribute__((ext_vector_type(8)))  __bf16 v8bf;
typedef __attribute__((ext_vector_type(8)))  float  v8f;

union FragU { v16bf v; v8bf h[2]; };

// 32-bit LDS byte address (VDST operand of async-to-LDS ops).
typedef const void __attribute__((address_space(3))) as3_void;
__device__ __forceinline__ unsigned lds_addr32(const void* p) {
    return (unsigned)(unsigned long long)(as3_void*)p;
}

// GLOBAL_LOAD_ASYNC_TO_LDS_B128: per-lane 16B global->LDS, tracked on ASYNCcnt.
__device__ __forceinline__ void async_copy_b128(unsigned lds_off, const void* gptr) {
    asm volatile("global_load_async_to_lds_b128 %0, %1, off"
                 :: "v"(lds_off), "v"(gptr) : "memory");
}
__device__ __forceinline__ void wait_async0() {
    asm volatile("s_wait_asynccnt 0x0" ::: "memory");
}

// Load one 16x32 bf16 WMMA fragment (A layout; B uses the mirrored layout with
// lane = column — same gather on a [16][K] k-major tile). Per ISA 7.12.2:
// lanes 0-15 hold K {0..7,16..23}, lanes 16-31 hold K {8..15,24..31};
// row/col = lane&15. Each half is one contiguous 16B LDS read.
__device__ __forceinline__ v16bf load_frag(const bf16* p, int stride) {
    const int L  = threadIdx.x & 31;
    const int r  = L & 15;
    const int kb = (L >> 4) << 3;
    FragU f;
    f.h[0] = *reinterpret_cast<const v8bf*>(p + (size_t)r * stride + kb);
    f.h[1] = *reinterpret_cast<const v8bf*>(p + (size_t)r * stride + kb + 16);
    return f.v;
}

__device__ __forceinline__ v8f wmma_bf16(v16bf a, v16bf b, v8f c) {
    return __builtin_amdgcn_wmma_f32_16x16x32_bf16(false, a, false, b,
                                                   (short)0, c, false, false);
}

// ---------------------------------------------------------------------------
// fp32 -> bf16 conversion
// ---------------------------------------------------------------------------
__global__ void cvt_f32_bf16(const float* __restrict__ s, bf16* __restrict__ d, int n) {
    int i = blockIdx.x * blockDim.x + threadIdx.x;
    if (i < n) d[i] = (bf16)s[i];
}

// ---------------------------------------------------------------------------
// C[M,N] = A[M,K] @ W[N,K]^T  (torch Linear).  Workgroup tile 256x64, 8 waves,
// each wave 32Mx64N (8 WMMA / 32-K slice).  K slices double-buffered in LDS,
// filled with async global->LDS copies overlapping the WMMA work.
// outf != nullptr -> f32 output + bias, else bf16 output.
// ---------------------------------------------------------------------------
__global__ __launch_bounds__(256)
void gemm_bf16(const bf16* __restrict__ A, const bf16* __restrict__ W,
               bf16* __restrict__ outb, float* __restrict__ outf,
               const float* __restrict__ bias, int M, int N, int K) {
    __shared__ __align__(16) bf16 Als[2][256 * 32];   // 2 x 16 KB
    __shared__ __align__(16) bf16 Bls[2][64 * 32];    // 2 x  4 KB
    const int tid  = threadIdx.x;
    const int lane = tid & 31;
    const int wave = tid >> 5;
    const int m0   = blockIdx.x * 256;
    const int n0   = blockIdx.y * 64;

    v8f acc0[4] = {};
    v8f acc1[4] = {};

    auto stage = [&](bf16* Ab, bf16* Bb, int k0) {
        // A tile 256x32: 1024 x 8-elem chunks, 4 per thread (async, no VGPR data)
#pragma unroll
        for (int i = 0; i < 4; ++i) {
            int id = tid + 256 * i;
            int row = id >> 2, off = (id & 3) << 3;
            async_copy_b128(lds_addr32(&Ab[row * 32 + off]),
                            &A[(size_t)(m0 + row) * K + k0 + off]);
        }
        // B tile 64x32 (k-major per output column): 1 chunk per thread
        int row = tid >> 2, off = (tid & 3) << 3;
        async_copy_b128(lds_addr32(&Bb[row * 32 + off]),
                        &W[(size_t)(n0 + row) * K + k0 + off]);
    };

    stage(Als[0], Bls[0], 0);
    wait_async0();
    __syncthreads();

    int cur = 0;
    for (int k0 = 0; k0 < K; k0 += 32) {
        if (k0 + 32 < K) stage(Als[cur ^ 1], Bls[cur ^ 1], k0 + 32);  // prefetch

        v16bf a0 = load_frag(Als[cur] + (wave * 32 + 0)  * 32, 32);
        v16bf a1 = load_frag(Als[cur] + (wave * 32 + 16) * 32, 32);
#pragma unroll
        for (int f = 0; f < 4; ++f) {
            v16bf b = load_frag(Bls[cur] + f * 16 * 32, 32);
            acc0[f] = wmma_bf16(a0, b, acc0[f]);
            acc1[f] = wmma_bf16(a1, b, acc1[f]);
        }

        wait_async0();       // this wave's prefetch landed
        __syncthreads();     // everyone done reading `cur`, prefetch visible
        cur ^= 1;
    }

    // C/D layout: VGPR r, lanes 0-15 -> M=r, lanes 16-31 -> M=8+r; N=lane&15.
    const int hi8   = (lane >> 4) << 3;
    const int ncol0 = n0 + (lane & 15);
#pragma unroll
    for (int f = 0; f < 4; ++f) {
        int col = ncol0 + f * 16;
#pragma unroll
        for (int r = 0; r < 8; ++r) {
            size_t i0 = (size_t)(m0 + wave * 32 + hi8 + r) * N + col;
            size_t i1 = i0 + (size_t)16 * N;
            if (outf) { outf[i0] = acc0[f][r] + bias[col];
                        outf[i1] = acc1[f][r] + bias[col]; }
            else      { outb[i0] = (bf16)acc0[f][r];
                        outb[i1] = (bf16)acc1[f][r]; }
        }
    }
}

// ---------------------------------------------------------------------------
// Windowed flash attention: 1 wave per (head, 16-query tile).
// scores = Q·K^T (no 1/sqrt(d) scaling, per reference), mask j>i || j<=i-256,
// online softmax, O = softmax(S)·V.
// ---------------------------------------------------------------------------
#define HSTRIDE 1024
#define HD      64
#define WIN     256

__global__ __launch_bounds__(32)
void attn_kernel(const bf16* __restrict__ Q, const bf16* __restrict__ Km,
                 const bf16* __restrict__ V, bf16* __restrict__ C) {
    __shared__ __align__(16) bf16 Kls[16 * 64];   // K tile, [key][d]
    __shared__ __align__(16) bf16 Pls[16 * 32];   // P tile, K padded 16->32
    __shared__ __align__(16) bf16 Vt [64 * 32];   // V^T tile, [d][key] padded
    const int lane = threadIdx.x;
    const int h    = blockIdx.y;
    const int q0   = blockIdx.x * 16;

    // Zero the K=16..31 pad regions once (never written afterwards).
    for (int id = lane; id < 256; id += 32)
        Pls[(id >> 4) * 32 + 16 + (id & 15)] = (bf16)0.0f;
    for (int id = lane; id < 1024; id += 32)
        Vt[(id >> 4) * 32 + 16 + (id & 15)] = (bf16)0.0f;

    // Q fragments (16 rows x hd=64 -> two 16x32 A fragments), loaded once.
    const bf16* qbase = Q + (size_t)q0 * HSTRIDE + h * HD;
    v16bf aq0 = load_frag(qbase,      HSTRIDE);
    v16bf aq1 = load_frag(qbase + 32, HSTRIDE);

    v8f o[4] = {};
    float mrun[8], lrun[8];
#pragma unroll
    for (int r = 0; r < 8; ++r) { mrun[r] = -1e30f; lrun[r] = 0.0f; }

    const int mloc = (lane >> 4) << 3;   // row offset of this half-wave
    const int nloc = lane & 15;          // column held by this lane

    int t_hi = q0 >> 4;
    int t_lo = t_hi - 16; if (t_lo < 0) t_lo = 0;

    for (int t = t_lo; t <= t_hi; ++t) {
        const int jt = t * 16;
        __syncthreads();
        // Stage K tile row-major and V tile transposed (128 x 8-elem chunks).
#pragma unroll
        for (int i = 0; i < 4; ++i) {
            int id   = lane + 32 * i;          // 0..127
            int krow = id >> 3;                // key 0..15
            int d0   = (id & 7) << 3;          // d   0..56
            const size_t g = (size_t)(jt + krow) * HSTRIDE + h * HD + d0;
            *reinterpret_cast<v8bf*>(&Kls[krow * 64 + d0]) =
                *reinterpret_cast<const v8bf*>(&Km[g]);
            v8bf vv = *reinterpret_cast<const v8bf*>(&V[g]);
#pragma unroll
            for (int e = 0; e < 8; ++e) Vt[(d0 + e) * 32 + krow] = vv[e];
        }
        __syncthreads();

        // S = Q·K^T (two chained WMMAs over hd=64)
        v16bf bk0 = load_frag(Kls,      64);
        v16bf bk1 = load_frag(Kls + 32, 64);
        v8f s = {};
        s = wmma_bf16(aq0, bk0, s);
        s = wmma_bf16(aq1, bk1, s);

        // Mask + online softmax (row spread over 16 lanes of a half-wave).
        float p[8], scale[8];
#pragma unroll
        for (int r = 0; r < 8; ++r) {
            int i = q0 + mloc + r;
            int j = jt + nloc;
            float sv = ((j > i) || (j + WIN <= i)) ? -1e30f : s[r];
            float mx = sv;
#pragma unroll
            for (int mm = 1; mm < 16; mm <<= 1)
                mx = fmaxf(mx, __shfl_xor(mx, mm, 32));
            float mnew = fmaxf(mrun[r], mx);
            scale[r] = __expf(mrun[r] - mnew);
            float pe = __expf(sv - mnew);
            float rs = pe;
#pragma unroll
            for (int mm = 1; mm < 16; mm <<= 1)
                rs += __shfl_xor(rs, mm, 32);
            lrun[r] = lrun[r] * scale[r] + rs;
            mrun[r] = mnew;
            p[r] = pe;
        }
#pragma unroll
        for (int f = 0; f < 4; ++f)
#pragma unroll
            for (int r = 0; r < 8; ++r) o[f][r] *= scale[r];

        // C-layout -> A-layout round trip through LDS for P.
#pragma unroll
        for (int r = 0; r < 8; ++r)
            Pls[(mloc + r) * 32 + nloc] = (bf16)p[r];
        __syncthreads();

        v16bf ap = load_frag(Pls, 32);     // K 16..31 are zeros
#pragma unroll
        for (int f = 0; f < 4; ++f) {
            v16bf bv = load_frag(Vt + f * 16 * 32, 32);
            o[f] = wmma_bf16(ap, bv, o[f]);
        }
    }

    // Normalize and write context (bf16, [S][H] with head-h columns).
#pragma unroll
    for (int r = 0; r < 8; ++r) {
        float inv = 1.0f / lrun[r];
        size_t row = (size_t)(q0 + mloc + r) * HSTRIDE + h * HD + nloc;
#pragma unroll
        for (int f = 0; f < 4; ++f)
            C[row + f * 16] = (bf16)(o[f][r] * inv);
    }
}

// ---------------------------------------------------------------------------
// Launch
// ---------------------------------------------------------------------------
extern "C" void kernel_launch(void* const* d_in, const int* in_sizes, int n_in,
                              void* d_out, int out_size, void* d_ws, size_t ws_size,
                              hipStream_t stream) {
    (void)in_sizes; (void)n_in; (void)out_size; (void)ws_size;
    const float* x  = (const float*)d_in[0];
    const float* wq = (const float*)d_in[1];
    const float* wk = (const float*)d_in[2];
    const float* wv = (const float*)d_in[3];
    const float* wo = (const float*)d_in[4];
    const float* bo = (const float*)d_in[5];

    const int S = 4096, H = 1024;
    const size_t nx = (size_t)S * H;   // 4.19M
    const size_t nw = (size_t)H * H;   // 1.05M

    char* w = (char*)d_ws;
    bf16* xb  = (bf16*)w; w += nx * 2;
    bf16* wqb = (bf16*)w; w += nw * 2;
    bf16* wkb = (bf16*)w; w += nw * 2;
    bf16* wvb = (bf16*)w; w += nw * 2;
    bf16* wob = (bf16*)w; w += nw * 2;
    bf16* Qb  = (bf16*)w; w += nx * 2;
    bf16* Kb  = (bf16*)w; w += nx * 2;
    bf16* Vb  = (bf16*)w; w += nx * 2;
    bf16* Cb  = (bf16*)w; w += nx * 2;

    cvt_f32_bf16<<<(int)((nx + 255) / 256), 256, 0, stream>>>(x,  xb,  (int)nx);
    cvt_f32_bf16<<<(int)((nw + 255) / 256), 256, 0, stream>>>(wq, wqb, (int)nw);
    cvt_f32_bf16<<<(int)((nw + 255) / 256), 256, 0, stream>>>(wk, wkb, (int)nw);
    cvt_f32_bf16<<<(int)((nw + 255) / 256), 256, 0, stream>>>(wv, wvb, (int)nw);
    cvt_f32_bf16<<<(int)((nw + 255) / 256), 256, 0, stream>>>(wo, wob, (int)nw);

    dim3 gg(S / 256, H / 64);   // 16 x 16 workgroups, 256 threads (8 waves)
    gemm_bf16<<<gg, 256, 0, stream>>>(xb, wqb, Qb, nullptr, nullptr, S, H, H);
    gemm_bf16<<<gg, 256, 0, stream>>>(xb, wkb, Kb, nullptr, nullptr, S, H, H);
    gemm_bf16<<<gg, 256, 0, stream>>>(xb, wvb, Vb, nullptr, nullptr, S, H, H);

    attn_kernel<<<dim3(S / 16, 16), 32, 0, stream>>>(Qb, Kb, Vb, Cb);

    gemm_bf16<<<gg, 256, 0, stream>>>(Cb, wob, nullptr, (float*)d_out, bo, S, H, H);
}
